// Block_75161927680501
// MI455X (gfx1250) — compile-verified, matrix operand order checked by reading, not run on hardware
//
#include <hip/hip_runtime.h>
#include <hip/hip_bf16.h>
#include <math.h>

typedef __attribute__((ext_vector_type(16))) __bf16 v16bf;
typedef __attribute__((ext_vector_type(8)))  __bf16 v8bf;
typedef __attribute__((ext_vector_type(4)))  __bf16 v4bf;
typedef __attribute__((ext_vector_type(8)))  float  v8f;

#define BATCH 4
#define SEQ   2048
#define CH    64
#define HSIZE 512
#define NH    8

__device__ __forceinline__ float geluf(float v) {
  return 0.5f * v * (1.0f + erff(v * 0.70710678118654752f));
}

// 16x32 bf16 WMMA A-operand (B uses the symmetric layout) from a row-major
// bf16 tile, as two contiguous 16-byte loads per lane. Per CDNA5 ISA 16-bit A
// map: lane L holds row (L&15); lanes 0-15 carry K {0..7,16..23}, lanes 16-31
// carry K {8..15,24..31}. All call sites keep p 16-byte aligned.
__device__ __forceinline__ v16bf load_tile_bf16(const __bf16* base, int ld, int lane) {
  int row  = lane & 15;
  int kOff = (lane >> 4) << 3;
  const __bf16* p = base + (size_t)row * ld + kOff;
  v8bf lo = *(const v8bf*)p;          // K = kOff   .. kOff+7
  v8bf hi = *(const v8bf*)(p + 16);   // K = kOff+16.. kOff+23
  return __builtin_shufflevector(lo, hi, 0, 1, 2, 3, 4, 5, 6, 7,
                                 8, 9, 10, 11, 12, 13, 14, 15);
}

__device__ __forceinline__ v8f wmma_bf16(v16bf a, v16bf b, v8f c) {
  return __builtin_amdgcn_wmma_f32_16x16x32_bf16(false, a, false, b, (short)0, c,
                                                 false, false);
}

// ---------------------------------------------------------------------------
// Kernel 0: one-shot weight transpose + bf16 convert. All WMMA B-operands are
// then contiguous bf16 tiles (row = output column, K contiguous).
// ---------------------------------------------------------------------------
__global__ void wprep_kernel(const float* __restrict__ Wqkv,
                             const float* __restrict__ Wout,
                             const float* __restrict__ W1,
                             const float* __restrict__ W2,
                             __bf16* __restrict__ wqkvT,
                             __bf16* __restrict__ woutT,
                             __bf16* __restrict__ w1T,
                             __bf16* __restrict__ w2T) {
  int i = blockIdx.x * 256 + threadIdx.x;
  const int nQKV = CH * 3 * HSIZE;   // 98304
  const int nOut = HSIZE * CH;       // 32768
  const int nM   = CH * CH;          // 4096
  if (i < nQKV) {
    int c = i / (3 * HSIZE), n = i - c * (3 * HSIZE);
    wqkvT[(size_t)n * CH + c] = (__bf16)Wqkv[i];
    return;
  }
  i -= nQKV;
  if (i < nOut) {
    int kk = i / CH, n = i - kk * CH;
    woutT[(size_t)n * HSIZE + kk] = (__bf16)Wout[i];
    return;
  }
  i -= nOut;
  if (i < nM) {
    int kk = i / CH, n = i - kk * CH;
    w1T[(size_t)n * CH + kk] = (__bf16)W1[i];
    return;
  }
  i -= nM;
  if (i < nM) {
    int kk = i / CH, n = i - kk * CH;
    w2T[(size_t)n * CH + kk] = (__bf16)W2[i];
  }
}

// ---------------------------------------------------------------------------
// Kernel 1: LN1 + fused QKV projection. 16 tokens per block; LN is computed by
// 16 lanes/token with shuffle reductions; 8 waves split the 1536 output
// columns (192 each), 2 WMMAs per 16-col tile. Q,K stored (B,H,N,D) bf16;
// V stored transposed (B,H,D,N) bf16 (so its store AND its attention loads
// are contiguous).
// ---------------------------------------------------------------------------
__global__ void ln_qkv_kernel(const float* __restrict__ x,
                              const float* __restrict__ g1,
                              const float* __restrict__ be1,
                              const __bf16* __restrict__ wqkvT,
                              const float* __restrict__ bqkv,
                              __bf16* __restrict__ q,
                              __bf16* __restrict__ k,
                              __bf16* __restrict__ vt) {
  __shared__ __bf16 ybuf[16][64];
  const int tid = threadIdx.x;
  const int tokBase = blockIdx.x * 16;
  {
    int t = tid >> 4;                 // token within tile (16 lanes each)
    int f = (tid & 15) << 2;          // 4 features per lane
    const float* xr = x + (size_t)(tokBase + t) * CH + f;
    float4 v4 = *(const float4*)xr;
    float s = v4.x + v4.y + v4.z + v4.w;
    s += __shfl_xor(s, 1, 32); s += __shfl_xor(s, 2, 32);
    s += __shfl_xor(s, 4, 32); s += __shfl_xor(s, 8, 32);
    float mean = s * (1.0f / CH);
    float dx = v4.x - mean, dy = v4.y - mean, dz = v4.z - mean, dw = v4.w - mean;
    float sq = dx * dx + dy * dy + dz * dz + dw * dw;
    sq += __shfl_xor(sq, 1, 32); sq += __shfl_xor(sq, 2, 32);
    sq += __shfl_xor(sq, 4, 32); sq += __shfl_xor(sq, 8, 32);
    float rstd = rsqrtf(sq * (1.0f / CH) + 1e-6f);
    v4bf o;
    o[0] = (__bf16)(dx * rstd * g1[f + 0] + be1[f + 0]);
    o[1] = (__bf16)(dy * rstd * g1[f + 1] + be1[f + 1]);
    o[2] = (__bf16)(dz * rstd * g1[f + 2] + be1[f + 2]);
    o[3] = (__bf16)(dw * rstd * g1[f + 3] + be1[f + 3]);
    *(v4bf*)&ybuf[t][f] = o;
  }
  __syncthreads();
  const int wv = tid >> 5, lane = tid & 31;
  const int hlf = lane >> 4;
  v16bf a0 = load_tile_bf16(&ybuf[0][0], 64, lane);        // K = 0..31
  v16bf a1 = load_tile_bf16(&ybuf[0][0] + 32, 64, lane);   // K = 32..63
  const int bb = tokBase >> 11;
  const int n0 = (tokBase & 2047) + hlf * 8;
  for (int j = 0; j < 12; ++j) {
    int colBase = wv * 192 + j * 16;
    v16bf b0 = load_tile_bf16(wqkvT + (size_t)colBase * CH, CH, lane);
    v16bf b1 = load_tile_bf16(wqkvT + (size_t)colBase * CH + 32, CH, lane);
    v8f acc = {};
    acc = wmma_bf16(a0, b0, acc);
    acc = wmma_bf16(a1, b1, acc);
    int col = colBase + (lane & 15);
    float bias = bqkv[col];
    int which = col >> 9;             // 0=Q 1=K 2=V
    int rem = col & 511;
    int hh = rem >> 6, dd = rem & 63;
    if (which == 2) {                 // V^T: tokens unit-stride -> one b128
      __bf16* dst = vt + ((size_t)(bb * NH + hh) * CH + dd) * SEQ + n0;
      v8bf o;
#pragma unroll
      for (int r = 0; r < 8; ++r) o[r] = (__bf16)(acc[r] + bias);
      *(v8bf*)dst = o;
    } else {                          // Q/K: (B,H,N,D), stride CH over tokens
      __bf16* dst = (which ? k : q) +
                    ((size_t)(bb * NH + hh) * SEQ + n0) * CH + dd;
#pragma unroll
      for (int r = 0; r < 8; ++r) dst[(size_t)r * CH] = (__bf16)(acc[r] + bias);
    }
  }
}

// ---------------------------------------------------------------------------
// Kernel 2: flash-style attention, softmax in the log2 domain (v_exp_f32 is
// base-2 on AMD: fold log2(e)/sqrt(d) into the score scale, saving a VALU mul
// per score element). One wave owns a 16-row Q tile; per 32-key chunk:
// 4 score WMMAs, online softmax (shuffle reductions inside 16-lane halves,
// matching the C-layout row map), P re-staged via 1KB/wave LDS to A-operand
// layout, 4 ctx WMMAs. Prefetch streams the next K/V chunk.
// ---------------------------------------------------------------------------
__global__ void attn_kernel(const __bf16* __restrict__ q,
                            const __bf16* __restrict__ k,
                            const __bf16* __restrict__ vt,
                            __bf16* __restrict__ ctx) {
  __shared__ __bf16 pbuf[8][16][32];
  const int tid = threadIdx.x, wv = tid >> 5, lane = tid & 31;
  const int hlf = lane >> 4, col = lane & 15;
  const int bh = blockIdx.x >> 4;
  const int qBase = ((((blockIdx.x & 15) << 3) + wv) << 4);
  const __bf16* qbh = q  + (size_t)bh * SEQ * CH;
  const __bf16* kbh = k  + (size_t)bh * SEQ * CH;
  const __bf16* vbh = vt + (size_t)bh * CH * SEQ;
  v16bf qa0 = load_tile_bf16(qbh + (size_t)qBase * CH,      CH, lane);
  v16bf qa1 = load_tile_bf16(qbh + (size_t)qBase * CH + 32, CH, lane);
  v8f acc0 = {}, acc1 = {}, acc2 = {}, acc3 = {};
  float m[8], l[8];
#pragma unroll
  for (int r = 0; r < 8; ++r) { m[r] = -3.0e38f; l[r] = 0.f; }
  const float kScale = 0.18033688011112043f;   // log2(e) / sqrt(64)

  for (int kc = 0; kc < SEQ; kc += 32) {
    if (kc + 32 < SEQ) {              // stream-ahead: lowers to global_prefetch
      __builtin_prefetch(kbh + (size_t)(kc + 32) * CH, 0, 3);
      __builtin_prefetch(vbh + kc + 32, 0, 3);
    }
    v8f s0 = {}, s1 = {};
    {
      v16bf kb0 = load_tile_bf16(kbh + (size_t)kc * CH,      CH, lane);
      v16bf kb1 = load_tile_bf16(kbh + (size_t)kc * CH + 32, CH, lane);
      s0 = wmma_bf16(qa0, kb0, s0);
      s0 = wmma_bf16(qa1, kb1, s0);
      kb0 = load_tile_bf16(kbh + (size_t)(kc + 16) * CH,      CH, lane);
      kb1 = load_tile_bf16(kbh + (size_t)(kc + 16) * CH + 32, CH, lane);
      s1 = wmma_bf16(qa0, kb0, s1);
      s1 = wmma_bf16(qa1, kb1, s1);
    }
    s0 *= kScale;                     // scores now in log2 domain
    s1 *= kScale;
#pragma unroll
    for (int r = 0; r < 8; ++r) {
      float mx = fmaxf(s0[r], s1[r]);
      mx = fmaxf(mx, __shfl_xor(mx, 1, 32));
      mx = fmaxf(mx, __shfl_xor(mx, 2, 32));
      mx = fmaxf(mx, __shfl_xor(mx, 4, 32));
      mx = fmaxf(mx, __shfl_xor(mx, 8, 32));
      float mn   = fmaxf(m[r], mx);
      float corr = exp2f(m[r] - mn);  // single v_exp_f32
      m[r] = mn;
      float p0 = exp2f(s0[r] - mn);
      float p1 = exp2f(s1[r] - mn);
      s0[r] = p0; s1[r] = p1;
      float ps = p0 + p1;
      ps += __shfl_xor(ps, 1, 32);
      ps += __shfl_xor(ps, 2, 32);
      ps += __shfl_xor(ps, 4, 32);
      ps += __shfl_xor(ps, 8, 32);
      l[r] = l[r] * corr + ps;
      acc0[r] *= corr; acc1[r] *= corr; acc2[r] *= corr; acc3[r] *= corr;
    }
    // C-layout -> A-operand layout via per-wave LDS tile
#pragma unroll
    for (int r = 0; r < 8; ++r) {
      pbuf[wv][r + hlf * 8][col]      = (__bf16)s0[r];
      pbuf[wv][r + hlf * 8][16 + col] = (__bf16)s1[r];
    }
    asm volatile("s_wait_dscnt 0" ::: "memory");
    v16bf pa = load_tile_bf16(&pbuf[wv][0][0], 32, lane);
    acc0 = wmma_bf16(pa, load_tile_bf16(vbh + (size_t)0  * SEQ + kc, SEQ, lane), acc0);
    acc1 = wmma_bf16(pa, load_tile_bf16(vbh + (size_t)16 * SEQ + kc, SEQ, lane), acc1);
    acc2 = wmma_bf16(pa, load_tile_bf16(vbh + (size_t)32 * SEQ + kc, SEQ, lane), acc2);
    acc3 = wmma_bf16(pa, load_tile_bf16(vbh + (size_t)48 * SEQ + kc, SEQ, lane), acc3);
  }
  const int bb = bh >> 3, hh = bh & 7;
#pragma unroll
  for (int r = 0; r < 8; ++r) {
    float inv = 1.0f / l[r];
    int nn = qBase + r + hlf * 8;
    size_t rowOff = ((size_t)bb * SEQ + nn) * HSIZE + hh * CH;
    ctx[rowOff + 0  + col] = (__bf16)(acc0[r] * inv);
    ctx[rowOff + 16 + col] = (__bf16)(acc1[r] * inv);
    ctx[rowOff + 32 + col] = (__bf16)(acc2[r] * inv);
    ctx[rowOff + 48 + col] = (__bf16)(acc3[r] * inv);
  }
}

// ---------------------------------------------------------------------------
// Kernel 3: out-proj (16x512 @ 512x64, 64 WMMAs) + GELU + residual,
// register-resident LN2 (shuffle reductions), MLP (2x 64x64 GEMMs, 16 WMMAs)
// + residual. One wave per 16-token tile, 8 tiles per block.
// ---------------------------------------------------------------------------
__global__ void proj_mlp_kernel(const float* __restrict__ x,
                                const __bf16* __restrict__ ctx,
                                const __bf16* __restrict__ woutT,
                                const float* __restrict__ bout,
                                const float* __restrict__ g2,
                                const float* __restrict__ be2,
                                const __bf16* __restrict__ w1T,
                                const float* __restrict__ b1,
                                const __bf16* __restrict__ w2T,
                                const float* __restrict__ b2,
                                float* __restrict__ out) {
  __shared__ __bf16 abuf[8][16][64];
  const int tid = threadIdx.x, wv = tid >> 5, lane = tid & 31;
  const int hlf = lane >> 4, col = lane & 15;
  const int tokBase = blockIdx.x * 128 + wv * 16;

  v8f acc[4] = {};
  for (int kc = 0; kc < HSIZE; kc += 32) {
    v16bf a = load_tile_bf16(ctx + (size_t)tokBase * HSIZE + kc, HSIZE, lane);
#pragma unroll
    for (int t = 0; t < 4; ++t)
      acc[t] = wmma_bf16(
          a, load_tile_bf16(woutT + (size_t)(t * 16) * HSIZE + kc, HSIZE, lane),
          acc[t]);
  }
  float x2[4][8];
#pragma unroll
  for (int t = 0; t < 4; ++t) {
    float bo = bout[t * 16 + col];
#pragma unroll
    for (int r = 0; r < 8; ++r) {
      int tok = tokBase + r + hlf * 8;
      x2[t][r] = geluf(acc[t][r] + bo) + x[(size_t)tok * CH + t * 16 + col];
    }
  }
  float mean[8], rstd[8];
#pragma unroll
  for (int r = 0; r < 8; ++r) {
    float s  = x2[0][r] + x2[1][r] + x2[2][r] + x2[3][r];
    float sq = x2[0][r] * x2[0][r] + x2[1][r] * x2[1][r] +
               x2[2][r] * x2[2][r] + x2[3][r] * x2[3][r];
    s  += __shfl_xor(s, 1, 32);  sq += __shfl_xor(sq, 1, 32);
    s  += __shfl_xor(s, 2, 32);  sq += __shfl_xor(sq, 2, 32);
    s  += __shfl_xor(s, 4, 32);  sq += __shfl_xor(sq, 4, 32);
    s  += __shfl_xor(s, 8, 32);  sq += __shfl_xor(sq, 8, 32);
    mean[r] = s * (1.0f / CH);
    float var = sq * (1.0f / CH) - mean[r] * mean[r];
    rstd[r] = rsqrtf(var + 1e-6f);
  }
#pragma unroll
  for (int t = 0; t < 4; ++t) {
    float gg = g2[t * 16 + col], bb = be2[t * 16 + col];
#pragma unroll
    for (int r = 0; r < 8; ++r)
      abuf[wv][r + hlf * 8][t * 16 + col] =
          (__bf16)((x2[t][r] - mean[r]) * rstd[r] * gg + bb);
  }
  asm volatile("s_wait_dscnt 0" ::: "memory");
  v16bf a0 = load_tile_bf16(&abuf[wv][0][0], 64, lane);
  v16bf a1 = load_tile_bf16(&abuf[wv][0][0] + 32, 64, lane);
  v8f hacc[4] = {};
#pragma unroll
  for (int t = 0; t < 4; ++t) {
    hacc[t] = wmma_bf16(a0, load_tile_bf16(w1T + (size_t)(t * 16) * CH, CH, lane), hacc[t]);
    hacc[t] = wmma_bf16(a1, load_tile_bf16(w1T + (size_t)(t * 16) * CH + 32, CH, lane), hacc[t]);
  }
#pragma unroll
  for (int t = 0; t < 4; ++t) {
    float bb = b1[t * 16 + col];
#pragma unroll
    for (int r = 0; r < 8; ++r)
      abuf[wv][r + hlf * 8][t * 16 + col] = (__bf16)geluf(hacc[t][r] + bb);
  }
  asm volatile("s_wait_dscnt 0" ::: "memory");
  a0 = load_tile_bf16(&abuf[wv][0][0], 64, lane);
  a1 = load_tile_bf16(&abuf[wv][0][0] + 32, 64, lane);
  v8f oacc[4] = {};
#pragma unroll
  for (int t = 0; t < 4; ++t) {
    oacc[t] = wmma_bf16(a0, load_tile_bf16(w2T + (size_t)(t * 16) * CH, CH, lane), oacc[t]);
    oacc[t] = wmma_bf16(a1, load_tile_bf16(w2T + (size_t)(t * 16) * CH + 32, CH, lane), oacc[t]);
  }
#pragma unroll
  for (int t = 0; t < 4; ++t) {
    float bb = b2[t * 16 + col];
#pragma unroll
    for (int r = 0; r < 8; ++r) {
      int tok = tokBase + r + hlf * 8;
      out[(size_t)tok * CH + t * 16 + col] = oacc[t][r] + bb + x2[t][r];
    }
  }
}

extern "C" void kernel_launch(void* const* d_in, const int* in_sizes, int n_in,
                              void* d_out, int out_size, void* d_ws, size_t ws_size,
                              hipStream_t stream) {
  const float* x    = (const float*)d_in[0];
  const float* g1   = (const float*)d_in[1];
  const float* be1  = (const float*)d_in[2];
  const float* Wqkv = (const float*)d_in[3];
  const float* bqkv = (const float*)d_in[4];
  const float* Wout = (const float*)d_in[5];
  const float* bout = (const float*)d_in[6];
  const float* g2   = (const float*)d_in[7];
  const float* be2  = (const float*)d_in[8];
  const float* W1   = (const float*)d_in[9];
  const float* b1   = (const float*)d_in[10];
  const float* W2   = (const float*)d_in[11];
  const float* b2   = (const float*)d_in[12];
  float* out = (float*)d_out;

  const size_t elems = (size_t)BATCH * NH * SEQ * CH;   // 4,194,304 per tensor
  __bf16* q     = (__bf16*)d_ws;          // Q  (B,H,N,D)   8 MB
  __bf16* k     = q  + elems;             // K  (B,H,N,D)   8 MB
  __bf16* vt    = k  + elems;             // V^T(B,H,D,N)   8 MB
  __bf16* ctx   = vt + elems;             // ctx(B,N,HS)    8 MB
  __bf16* wqkvT = ctx + elems;            // 1536x64
  __bf16* woutT = wqkvT + (size_t)CH * 3 * HSIZE;   // 64x512
  __bf16* w1T   = woutT + (size_t)HSIZE * CH;       // 64x64
  __bf16* w2T   = w1T + (size_t)CH * CH;            // 64x64

  const int wtot = CH * 3 * HSIZE + HSIZE * CH + 2 * CH * CH;  // 139264
  wprep_kernel<<<(wtot + 255) / 256, 256, 0, stream>>>(Wqkv, Wout, W1, W2,
                                                       wqkvT, woutT, w1T, w2T);
  ln_qkv_kernel<<<(BATCH * SEQ) / 16, 256, 0, stream>>>(x, g1, be1, wqkvT, bqkv,
                                                        q, k, vt);
  attn_kernel<<<BATCH * NH * (SEQ / 128), 256, 0, stream>>>(q, k, vt, ctx);
  proj_mlp_kernel<<<(BATCH * SEQ) / 128, 256, 0, stream>>>(x, ctx, woutT, bout,
                                                           g2, be2, w1T, b1,
                                                           w2T, b2, out);
}
